// RWKV_Tmix_x060_2911987827243
// MI455X (gfx1250) — compile-verified, matrix operand order checked by reading, not run on hardware
//
#include <hip/hip_runtime.h>
#include <hip/hip_bf16.h>
#include <stdint.h>

// ---------------- types ----------------
typedef __attribute__((ext_vector_type(16))) __bf16 v16bf;
typedef __attribute__((ext_vector_type(8)))  __bf16 v8bf;
typedef __attribute__((ext_vector_type(8)))  float  v8f;
typedef unsigned int u32x4 __attribute__((ext_vector_type(4)));
typedef int          i32x8 __attribute__((ext_vector_type(8)));
typedef int          i32x4 __attribute__((ext_vector_type(4)));

#define BB 4
#define TT 1024
#define CC 1024
#define HH 16
#define NN 64
#define BT (BB*TT)          // 4096 tokens
#define LN_EPS 6.4e-4f      // 1e-5 * 8^2
#define CHK 16              // timesteps per TDM-staged chunk in the WKV scan

#if defined(__has_builtin)
#if __has_builtin(__builtin_amdgcn_tensor_load_to_lds) && __has_builtin(__builtin_amdgcn_s_wait_tensorcnt)
#define HAVE_TDM 1
#endif
#endif

__device__ __forceinline__ __bf16 f2bf(float f) {
  union { float f; uint32_t u; } v; v.f = f;
  uint32_t r = v.u + 0x7FFFu + ((v.u >> 16) & 1u);   // round-to-nearest-even
  uint16_t h = (uint16_t)(r >> 16);
  __bf16 o; __builtin_memcpy(&o, &h, sizeof(o));
  return o;
}

// CDNA5 16-bit A/B fragment: lane half splits K into 8-elem runs at +0/+16.
__device__ __forceinline__ v16bf load_frag(const __bf16* __restrict__ p) {
  v8bf lo = *(const v8bf*)(p);
  v8bf hi = *(const v8bf*)(p + 16);
  return __builtin_shufflevector(lo, hi, 0,1,2,3,4,5,6,7,8,9,10,11,12,13,14,15);
}

// ---------------- generic bf16 WMMA GEMM, register-blocked MWx(16) x NWx(16) ----------------
// A: [M][K] bf16 (lda); Bt: [N][K] bf16 transposed weight (ldb); K % 32 == 0.
// EP: 0 = f32 store; 1 = tanh->bf16; 2 = mix->bf16 (ep0 + ep1*(ep2[col]+acc)); 3 = acc+ep2[col]->f32
template<int MW, int NW, int EP>
__global__ void gemm_wmma(const __bf16* __restrict__ A, int lda,
                          const __bf16* __restrict__ Bt, int ldb,
                          int M, int N, int K,
                          void* __restrict__ Out, int ldo,
                          const float* __restrict__ ep0,
                          const float* __restrict__ ep1,
                          const float* __restrict__ ep2) {
  const int lane = threadIdx.x & 31;
  const int wid  = blockIdx.x * (blockDim.x >> 5) + (threadIdx.x >> 5);
  const int ntm  = M / (16 * MW);
  const int ntn  = N / (16 * NW);
  if (wid >= ntm * ntn) return;           // wave-uniform: EXEC all-ones for WMMA
  const int tm   = wid / ntn;
  const int tn   = wid % ntn;
  const int lm   = lane & 15;
  const int half = lane >> 4;

  const __bf16* pa = A  + (size_t)(tm * MW * 16 + lm) * lda + half * 8;
  const __bf16* pb = Bt + (size_t)(tn * NW * 16 + lm) * ldb + half * 8;

  v8f acc[MW][NW] = {};
  for (int k = 0; k < K; k += 32) {
    v16bf af[MW];
#pragma unroll
    for (int mi = 0; mi < MW; ++mi) af[mi] = load_frag(pa + (size_t)mi * 16 * lda + k);
#pragma unroll
    for (int f = 0; f < NW; ++f) {
      v16bf bfrag = load_frag(pb + (size_t)f * 16 * ldb + k);
#pragma unroll
      for (int mi = 0; mi < MW; ++mi)
        acc[mi][f] = __builtin_amdgcn_wmma_f32_16x16x32_bf16(
                         false, af[mi], false, bfrag, (short)0, acc[mi][f], false, false);
    }
  }

#pragma unroll
  for (int mi = 0; mi < MW; ++mi) {
#pragma unroll
    for (int f = 0; f < NW; ++f) {
      const int col = (tn * NW + f) * 16 + lm;
#pragma unroll
      for (int e = 0; e < 8; ++e) {
        const int row = (tm * MW + mi) * 16 + half * 8 + e;  // C/D: VGPR e -> M=e (+8 upper half)
        const float va = acc[mi][f][e];
        const size_t o = (size_t)row * ldo + col;
        if (EP == 0) {
          ((float*)Out)[o] = va;
        } else if (EP == 1) {
          ((__bf16*)Out)[o] = f2bf(tanhf(va));
        } else if (EP == 2) {
          const size_t xi = (size_t)row * CC + col;
          ((__bf16*)Out)[o] = f2bf(ep0[xi] + ep1[xi] * (ep2[col] + va));
        } else {
          ((float*)Out)[o] = va + ep2[col];
        }
      }
    }
  }
}

// ---------------- prep kernels ----------------
__global__ void conv_t(const float* __restrict__ in, __bf16* __restrict__ out, int K, int N) {
  size_t idx = (size_t)blockIdx.x * blockDim.x + threadIdx.x;
  if (idx >= (size_t)K * N) return;
  int k = (int)(idx / N), n = (int)(idx % N);
  out[(size_t)n * K + k] = f2bf(in[idx]);
}

__global__ void pack_maa(const float* __restrict__ w, const float* __restrict__ k,
                         const float* __restrict__ v, const float* __restrict__ r,
                         const float* __restrict__ g, float* __restrict__ out) {
  int c = blockIdx.x * blockDim.x + threadIdx.x;
  if (c < CC) {
    out[c] = w[c]; out[CC + c] = k[c]; out[2*CC + c] = v[c];
    out[3*CC + c] = r[c]; out[4*CC + c] = g[c];
  }
}

__global__ void shift_kernel(const float* __restrict__ x, const float* __restrict__ maa_x,
                             float* __restrict__ xx, __bf16* __restrict__ xxxb) {
  size_t idx = (size_t)blockIdx.x * blockDim.x + threadIdx.x;
  if (idx >= (size_t)BT * CC) return;
  int c  = (int)(idx & (CC - 1));
  int bt = (int)(idx >> 10);
  int t  = bt & (TT - 1);
  float xv = x[idx];
  float pv = (t == 0) ? 0.f : x[idx - CC];
  float d  = pv - xv;
  xx[idx]   = d;
  xxxb[idx] = f2bf(xv + d * maa_x[c]);
}

// ---------------- Tensor Data Mover: 2D tile load (64 f32 wide x CHK rows, stride 1024 f32) ----
#ifdef HAVE_TDM
__device__ __forceinline__ void tdm_load_2d(unsigned lds_off, const float* gsrc) {
  unsigned long long ga = (unsigned long long)(uintptr_t)gsrc;
  u32x4 g0;
  g0[0] = 1u;                                   // count=1 (valid user descriptor)
  g0[1] = lds_off;                              // lds_addr (bytes)
  g0[2] = (unsigned)ga;                         // global_addr[31:0]
  g0[3] = (unsigned)(ga >> 32) | (2u << 30);    // global_addr[56:32] | type=2 ("image")
  i32x8 g1;
  g1[0] = (int)(2u << 16);                      // workgroup_mask=0, data_size=2 (4B)
  g1[1] = (int)(64u << 16);                     // tensor_dim0 = 64 (low half)
  g1[2] = (int)((unsigned)CHK << 16);           // tensor_dim0 hi=0 | tensor_dim1 = CHK
  g1[3] = (int)(64u << 16);                     // tensor_dim1 hi=0 | tile_dim0 = 64
  g1[4] = (int)CHK;                             // tile_dim1 = CHK, tile_dim2 = 0
  g1[5] = 1024;                                 // tensor_dim0_stride = 1024 elements
  g1[6] = 0;
  g1[7] = 0;
  i32x4 z4 = (i32x4)0;
#if defined(__clang_major__) && (__clang_major__ >= 23)
  i32x8 z8 = (i32x8)0;
  __builtin_amdgcn_tensor_load_to_lds(g0, g1, z4, z4, z8, 0);
#else
  __builtin_amdgcn_tensor_load_to_lds(g0, g1, z4, z4, 0);
#endif
}
#endif

// ---------------- WKV6 sequential scan with TDM double-buffered staging ----------------
// grid = B*H blocks, 64 threads (2 waves); thread i owns state column S[:,i] in VGPRs.
__global__ void wkv6_kernel(const float* __restrict__ r, const float* __restrict__ k,
                            const float* __restrict__ v, const float* __restrict__ w,
                            const float* __restrict__ u, float* __restrict__ y) {
  const int bh = blockIdx.x;
  const int b = bh >> 4, h = bh & 15;
  const int i = threadIdx.x;
  __shared__ float rb[2][CHK * NN];
  __shared__ float kb[2][CHK * NN];
  __shared__ float wb[2][CHK * NN];
  __shared__ float vb[2][CHK * NN];
  __shared__ float us[NN];
  float state[NN];
#pragma unroll
  for (int j = 0; j < NN; ++j) state[j] = 0.f;
  us[i] = u[h * NN + i];

  const size_t hoff = (size_t)b * TT * CC + h * NN;
  const int nchunks = TT / CHK;

#ifdef HAVE_TDM
  const unsigned rb0 = (unsigned)(uintptr_t)&rb[0][0];
  const unsigned kb0 = (unsigned)(uintptr_t)&kb[0][0];
  const unsigned wb0 = (unsigned)(uintptr_t)&wb[0][0];
  const unsigned vb0 = (unsigned)(uintptr_t)&vb[0][0];
  const unsigned bufB = CHK * NN * 4;           // bytes per buffer
  if (i < 32) {                                  // wave 0 issues DMAs
    tdm_load_2d(rb0, r + hoff);
    tdm_load_2d(kb0, k + hoff);
    tdm_load_2d(wb0, w + hoff);
    tdm_load_2d(vb0, v + hoff);
  }
#endif

  for (int c = 0; c < nchunks; ++c) {
    const int buf = c & 1;
#ifdef HAVE_TDM
    if (i < 32 && (c + 1) < nchunks) {           // prefetch next chunk into other buffer
      const unsigned nb = (unsigned)((c + 1) & 1) * bufB;
      const size_t go = hoff + (size_t)(c + 1) * CHK * CC;
      tdm_load_2d(rb0 + nb, r + go);
      tdm_load_2d(kb0 + nb, k + go);
      tdm_load_2d(wb0 + nb, w + go);
      tdm_load_2d(vb0 + nb, v + go);
    }
    if ((c + 1) < nchunks) __builtin_amdgcn_s_wait_tensorcnt(4);
    else                   __builtin_amdgcn_s_wait_tensorcnt(0);
#else
    {
      const size_t go = hoff + (size_t)c * CHK * CC;
      for (int tt = 0; tt < CHK; ++tt) {
        rb[buf][tt * NN + i] = r[go + (size_t)tt * CC + i];
        kb[buf][tt * NN + i] = k[go + (size_t)tt * CC + i];
        wb[buf][tt * NN + i] = w[go + (size_t)tt * CC + i];
        vb[buf][tt * NN + i] = v[go + (size_t)tt * CC + i];
      }
    }
#endif
    __syncthreads();
    // convert raw w -> decay = exp(-exp(w)) in place (one exp pair per element total)
    for (int tt = 0; tt < CHK; ++tt)
      wb[buf][tt * NN + i] = __expf(-__expf(wb[buf][tt * NN + i]));
    __syncthreads();

    for (int tt = 0; tt < CHK; ++tt) {
      const float vv = vb[buf][tt * NN + i];
      float yy = 0.f;
#pragma unroll
      for (int j = 0; j < NN; ++j) {
        const float kv = kb[buf][tt * NN + j] * vv;
        yy += rb[buf][tt * NN + j] * (state[j] + us[j] * kv);
        state[j] = state[j] * wb[buf][tt * NN + j] + kv;
      }
      y[hoff + (size_t)(c * CHK + tt) * CC + i] = yy;
    }
    __syncthreads();   // all reads done before this buffer is DMA-overwritten (chunk c+2)
  }
}

// ---------------- GroupNorm(n_head groups) + SiLU gate -> bf16 ----------------
__global__ void gn_gate_kernel(const float* __restrict__ y, const float* __restrict__ graw,
                               const float* __restrict__ lnw, const float* __restrict__ lnb,
                               __bf16* __restrict__ out) {
  const int tid  = threadIdx.x;          // 256 threads = 4 groups of 64
  const int grp  = tid >> 6;
  const int lane = tid & 63;
  const int gidx = blockIdx.x * 4 + grp; // bt*16 + h
  const int bt = gidx >> 4, h = gidx & 15;
  const size_t base = (size_t)bt * CC + h * NN;
  __shared__ float red[256];
  const float val = y[base + lane];
  red[tid] = val; __syncthreads();
  for (int s = 32; s > 0; s >>= 1) { if (lane < s) red[tid] += red[tid + s]; __syncthreads(); }
  const float mu = red[grp * 64] * (1.f / NN);
  __syncthreads();
  const float d = val - mu;
  red[tid] = d * d; __syncthreads();
  for (int s = 32; s > 0; s >>= 1) { if (lane < s) red[tid] += red[tid + s]; __syncthreads(); }
  const float var = red[grp * 64] * (1.f / NN);
  const int c = h * NN + lane;
  const float yn = d * rsqrtf(var + LN_EPS) * lnw[c] + lnb[c];
  const float gx = graw[base + lane];
  const float gs = gx / (1.f + __expf(-gx));     // SiLU
  out[base + lane] = f2bf(yn * gs);
}

// ---------------- host launcher ----------------
extern "C" void kernel_launch(void* const* d_in, const int* in_sizes, int n_in,
                              void* d_out, int out_size, void* d_ws, size_t ws_size,
                              hipStream_t stream) {
  const float* x      = (const float*)d_in[0];
  const float* maa_x  = (const float*)d_in[1];
  const float* maa_w  = (const float*)d_in[2];
  const float* maa_k  = (const float*)d_in[3];
  const float* maa_v  = (const float*)d_in[4];
  const float* maa_r  = (const float*)d_in[5];
  const float* maa_g  = (const float*)d_in[6];
  const float* maa_w1 = (const float*)d_in[7];    // [1024,160]
  const float* maa_w2 = (const float*)d_in[8];    // [5,32,1024]
  const float* tdecay = (const float*)d_in[9];    // [1024]
  const float* dw1    = (const float*)d_in[10];   // [1024,64]
  const float* dw2    = (const float*)d_in[11];   // [64,1024]
  const float* faaaa  = (const float*)d_in[12];   // [16,64]
  const float* Wr     = (const float*)d_in[13];
  const float* Wk     = (const float*)d_in[14];
  const float* Wv     = (const float*)d_in[15];
  const float* Wg     = (const float*)d_in[16];
  const float* Wo     = (const float*)d_in[17];
  const float* lnw    = (const float*)d_in[18];
  const float* lnb    = (const float*)d_in[19];
  float* out = (float*)d_out;

  char* ws = (char*)d_ws;
  size_t off = 0;
  auto alloc = [&](size_t bytes) -> void* {
    off = (off + 255) & ~(size_t)255;
    void* p = ws + off; off += bytes; return p;
  };

  __bf16* WrT  = (__bf16*)alloc((size_t)CC * CC * 2);
  __bf16* WkT  = (__bf16*)alloc((size_t)CC * CC * 2);
  __bf16* WvT  = (__bf16*)alloc((size_t)CC * CC * 2);
  __bf16* WgT  = (__bf16*)alloc((size_t)CC * CC * 2);
  __bf16* WoT  = (__bf16*)alloc((size_t)CC * CC * 2);
  __bf16* w1T  = (__bf16*)alloc((size_t)160 * CC * 2);
  __bf16* w2T  = (__bf16*)alloc((size_t)5 * CC * 32 * 2);
  __bf16* dw1T = (__bf16*)alloc((size_t)64 * CC * 2);
  __bf16* dw2T = (__bf16*)alloc((size_t)CC * 64 * 2);
  float*  maa5 = (float*)alloc((size_t)5 * CC * 4);
  float*  xxf  = (float*)alloc((size_t)BT * CC * 4);    // reused as y
  __bf16* xxxb = (__bf16*)alloc((size_t)BT * CC * 2);   // reused as gated-bf16
  __bf16* mixb = (__bf16*)alloc((size_t)BT * 160 * 2);
  __bf16* xsb  = (__bf16*)alloc((size_t)5 * BT * CC * 2);
  __bf16* dtmp = (__bf16*)alloc((size_t)BT * 64 * 2);
  float*  rf   = (float*)alloc((size_t)BT * CC * 4);
  float*  kf   = (float*)alloc((size_t)BT * CC * 4);
  float*  vf   = (float*)alloc((size_t)BT * CC * 4);
  float*  wf   = (float*)alloc((size_t)BT * CC * 4);
  float*  gf   = (float*)alloc((size_t)BT * CC * 4);
  float*  yf   = xxf;
  __bf16* yngb = xxxb;

  auto cdiv = [](int a, int b) { return (a + b - 1) / b; };

  conv_t<<<cdiv(CC * CC, 256), 256, 0, stream>>>(Wr, WrT, CC, CC);
  conv_t<<<cdiv(CC * CC, 256), 256, 0, stream>>>(Wk, WkT, CC, CC);
  conv_t<<<cdiv(CC * CC, 256), 256, 0, stream>>>(Wv, WvT, CC, CC);
  conv_t<<<cdiv(CC * CC, 256), 256, 0, stream>>>(Wg, WgT, CC, CC);
  conv_t<<<cdiv(CC * CC, 256), 256, 0, stream>>>(Wo, WoT, CC, CC);
  conv_t<<<cdiv(CC * 160, 256), 256, 0, stream>>>(maa_w1, w1T, CC, 160);
  for (int s = 0; s < 5; ++s)
    conv_t<<<cdiv(32 * CC, 256), 256, 0, stream>>>(maa_w2 + (size_t)s * 32 * CC,
                                                   w2T + (size_t)s * CC * 32, 32, CC);
  conv_t<<<cdiv(CC * 64, 256), 256, 0, stream>>>(dw1, dw1T, CC, 64);
  conv_t<<<cdiv(64 * CC, 256), 256, 0, stream>>>(dw2, dw2T, 64, CC);
  pack_maa<<<cdiv(CC, 256), 256, 0, stream>>>(maa_w, maa_k, maa_v, maa_r, maa_g, maa5);

  shift_kernel<<<cdiv(BT * CC, 256), 256, 0, stream>>>(x, maa_x, xxf, xxxb);

  // G1: mixlora = tanh(xxx @ maa_w1)  [4096,160] -> bf16   (MW=2, NW=2)
  gemm_wmma<2, 2, 1><<<cdiv((BT / 32) * (160 / 32), 4), 128, 0, stream>>>(
      xxxb, CC, w1T, CC, BT, 160, 1024, mixb, 160, nullptr, nullptr, nullptr);

  // G2 x5: xs = bf16( x + xx*(maa_s + mix_s @ w2_s) )   K=32  (MW=2, NW=4)
  for (int s = 0; s < 5; ++s)
    gemm_wmma<2, 4, 2><<<cdiv((BT / 32) * (CC / 64), 4), 128, 0, stream>>>(
        mixb + s * 32, 160, w2T + (size_t)s * CC * 32, 32, BT, CC, 32,
        xsb + (size_t)s * BT * CC, CC, x, xxf, maa5 + s * CC);

  // G3: r/k/v/g projections  (xs order: 0=w,1=k,2=v,3=r,4=g)
  {
    int blocks = cdiv((BT / 32) * (CC / 64), 4);
    gemm_wmma<2, 4, 0><<<blocks, 128, 0, stream>>>(xsb + (size_t)3 * BT * CC, CC, WrT, CC,
        BT, CC, CC, rf, CC, nullptr, nullptr, nullptr);
    gemm_wmma<2, 4, 0><<<blocks, 128, 0, stream>>>(xsb + (size_t)1 * BT * CC, CC, WkT, CC,
        BT, CC, CC, kf, CC, nullptr, nullptr, nullptr);
    gemm_wmma<2, 4, 0><<<blocks, 128, 0, stream>>>(xsb + (size_t)2 * BT * CC, CC, WvT, CC,
        BT, CC, CC, vf, CC, nullptr, nullptr, nullptr);
    gemm_wmma<2, 4, 0><<<blocks, 128, 0, stream>>>(xsb + (size_t)4 * BT * CC, CC, WgT, CC,
        BT, CC, CC, gf, CC, nullptr, nullptr, nullptr);
  }
  // G4: dtmp = tanh(xw @ decay_w1)  [4096,64] -> bf16
  gemm_wmma<2, 4, 1><<<cdiv((BT / 32) * (64 / 64), 4), 128, 0, stream>>>(
      xsb, CC, dw1T, CC, BT, 64, 1024, dtmp, 64, nullptr, nullptr, nullptr);
  // G5: w = time_decay + dtmp @ decay_w2   K=64
  gemm_wmma<2, 4, 3><<<cdiv((BT / 32) * (CC / 64), 4), 128, 0, stream>>>(
      dtmp, 64, dw2T, 64, BT, CC, 64, wf, CC, nullptr, nullptr, tdecay);

  // WKV6 scan (TDM double-buffered)
  wkv6_kernel<<<BB * HH, NN, 0, stream>>>(rf, kf, vf, wf, faaaa, yf);

  // GroupNorm + SiLU gate -> bf16
  gn_gate_kernel<<<(BT * HH) / 4, 256, 0, stream>>>(yf, gf, lnw, lnb, yngb);

  // G6: out = (yn*g) @ Wo  -> f32
  gemm_wmma<2, 4, 0><<<cdiv((BT / 32) * (CC / 64), 4), 128, 0, stream>>>(
      yngb, CC, WoT, CC, BT, CC, CC, out, CC, nullptr, nullptr, nullptr);
}
